// MultiheadLatentAttention_46935402611124
// MI455X (gfx1250) — matrix operand order repaired and checked by reading, not compiled, not run on hardware
//
#include <hip/hip_runtime.h>
#include <hip/hip_bf16.h>

typedef __bf16 v16bf __attribute__((ext_vector_type(16)));
typedef float  v8f   __attribute__((ext_vector_type(8)));

#define B_   4
#define S_   2048
#define E_   1024
#define H_   16
#define KVH_ 4
#define D_   64
#define QL_  64
#define KVL_ 32
#define BS_  (B_ * S_)
#define LN10000 9.210340372f

#define WMMA_BF16(a, b, c) \
  __builtin_amdgcn_wmma_f32_16x16x32_bf16(false, (a), false, (b), (short)0, (c), false, false)

// ---------------- fragment loaders (wave32 layouts from CDNA5 ISA) ---------------
// A (16x32 bf16): lane 0-15 -> row M=lane, K in {0..7}+{16..23}; lane 16-31 -> M=lane-16,
// K in {8..15}+{24..31}. Two contiguous 8-element runs -> two 16B loads.
static __device__ __forceinline__ v16bf load_a_frag(const __bf16* A, int lda, int m0, int k0) {
  int lane = threadIdx.x & 31;
  int half = lane >> 4, mm = lane & 15;
  const __bf16* p = A + (size_t)(m0 + mm) * lda + k0 + half * 8;
  union { v16bf v; uint4 q[2]; } r;
  r.q[0] = *(const uint4*)(p);
  r.q[1] = *(const uint4*)(p + 16);
  return r.v;
}
// B (32x16 bf16) from a transposed [N][K] matrix: lane holds column N=lane&15,
// K in {0..7}+{16..23} (lanes 0-15) or {8..15}+{24..31} (lanes 16-31), matching A.
static __device__ __forceinline__ v16bf load_bT_frag(const __bf16* BT, int ldb, int n0, int k0) {
  int lane = threadIdx.x & 31;
  int half = lane >> 4, nn = lane & 15;
  const __bf16* p = BT + (size_t)(n0 + nn) * ldb + k0 + half * 8;
  union { v16bf v; uint4 q[2]; } r;
  r.q[0] = *(const uint4*)(p);
  r.q[1] = *(const uint4*)(p + 16);
  return r.v;
}

// ---------------- casts --------------------------------------------------------------------
__global__ void k_cast8(const float* __restrict__ in, __bf16* __restrict__ out, int n8) {
  int i = blockIdx.x * 256 + threadIdx.x;   // 8 elements per thread
  if (i < n8) {
    const float4* p = (const float4*)in + (size_t)i * 2;
    float4 x0 = p[0], x1 = p[1];
    union { __bf16 h[8]; uint4 q; } r;
    r.h[0] = (__bf16)x0.x; r.h[1] = (__bf16)x0.y; r.h[2] = (__bf16)x0.z; r.h[3] = (__bf16)x0.w;
    r.h[4] = (__bf16)x1.x; r.h[5] = (__bf16)x1.y; r.h[6] = (__bf16)x1.z; r.h[7] = (__bf16)x1.w;
    ((uint4*)out)[i] = r.q;
  }
}
__global__ void k_cast_transpose(const float* in, __bf16* out, int rows, int cols) {
  int i = blockIdx.x * 256 + threadIdx.x;
  if (i < rows * cols) {
    int r = i / cols, c = i % cols;
    out[(size_t)c * rows + r] = (__bf16)in[i];
  }
}

// ---------------- 1) lat = x @ [Wqa | Wkva]  (8192 x 1024) x (1024 x 96) -------------------
// 128-row blocks, each wave owns a 32-row strip: 2 A-frags x 6 B-frags -> 12 WMMA / K-step.
// A prefetched one K-step ahead; all 6 B fragments loaded before the WMMA batch.
__global__ __launch_bounds__(128) void k_lat(const __bf16* __restrict__ xb,
                                             const __bf16* __restrict__ wqaT,
                                             const __bf16* __restrict__ wkvaT,
                                             __bf16* __restrict__ lat) {
  int w = threadIdx.x >> 5;
  int m0 = blockIdx.x * 128 + w * 32;
  v8f c[12] = {};
  v16bf a[2], an[2], b[6];
  a[0] = load_a_frag(xb, E_, m0, 0);
  a[1] = load_a_frag(xb, E_, m0 + 16, 0);
  for (int k = 0; k < E_; k += 32) {
#pragma unroll
    for (int nt = 0; nt < 6; ++nt) {
      const __bf16* BT = (nt < 4) ? wqaT : wkvaT;
      int n0 = (nt < 4) ? nt * 16 : (nt - 4) * 16;
      b[nt] = load_bT_frag(BT, E_, n0, k);
    }
    if (k + 32 < E_) {
      an[0] = load_a_frag(xb, E_, m0, k + 32);
      an[1] = load_a_frag(xb, E_, m0 + 16, k + 32);
    } else { an[0] = a[0]; an[1] = a[1]; }
#pragma unroll
    for (int mi = 0; mi < 2; ++mi)
#pragma unroll
      for (int nt = 0; nt < 6; ++nt)
        c[mi * 6 + nt] = WMMA_BF16(a[mi], b[nt], c[mi * 6 + nt]);
    a[0] = an[0]; a[1] = an[1];
  }
  int lane = threadIdx.x & 31, half = lane >> 4, nm = lane & 15;
#pragma unroll
  for (int mi = 0; mi < 2; ++mi)
#pragma unroll
    for (int nt = 0; nt < 6; ++nt) {
      union { v8f v; float f[8]; } cc; cc.v = c[mi * 6 + nt];
#pragma unroll
      for (int j = 0; j < 8; ++j) {
        int row = m0 + mi * 16 + j + half * 8;
        lat[(size_t)row * 96 + nt * 16 + nm] = (__bf16)cc.f[j];
      }
    }
}

// ---------------- 2) q = lat[:, :64] @ Wqb, fused RoPE (+ softmax scale), [b][h][s][d] -----
__global__ __launch_bounds__(128) void k_qproj(const __bf16* __restrict__ lat,
                                               const __bf16* __restrict__ wqbT,
                                               __bf16* __restrict__ qh) {
  int w = threadIdx.x >> 5;
  int m0 = blockIdx.x * 128 + w * 32;
  int h  = blockIdx.y;
  int n0g = h * 64;
  v8f c[8] = {};
  v16bf a[2][2], b[2][4];
#pragma unroll
  for (int ks = 0; ks < 2; ++ks) {
    a[ks][0] = load_a_frag(lat, 96, m0, ks * 32);
    a[ks][1] = load_a_frag(lat, 96, m0 + 16, ks * 32);
#pragma unroll
    for (int nt = 0; nt < 4; ++nt)
      b[ks][nt] = load_bT_frag(wqbT, QL_, n0g + nt * 16, ks * 32);
  }
#pragma unroll
  for (int ks = 0; ks < 2; ++ks)
#pragma unroll
    for (int mi = 0; mi < 2; ++mi)
#pragma unroll
      for (int nt = 0; nt < 4; ++nt)
        c[mi * 4 + nt] = WMMA_BF16(a[ks][mi], b[ks][nt], c[mi * 4 + nt]);

  int lane = threadIdx.x & 31, half = lane >> 4, nm = lane & 15;
#pragma unroll
  for (int mi = 0; mi < 2; ++mi)
#pragma unroll
    for (int nt = 0; nt < 4; ++nt) {
      union { v8f v; float f[8]; } cc; cc.v = c[mi * 4 + nt];
#pragma unroll
      for (int j = 0; j < 8; ++j) {
        float val = cc.f[j];
        float partner = __shfl_xor(val, 1, 32); // RoPE pair = adjacent column = adjacent lane
        int d = nt * 16 + nm;
        int jf = d >> 1;
        int row = m0 + mi * 16 + j + half * 8;
        int s = row & (S_ - 1), bb = row >> 11;
        float theta = (float)s * __expf(-(float)jf * (LN10000 / 32.0f));
        float cs = __cosf(theta), sn = __sinf(theta);
        float r = (d & 1) ? fmaf(partner, sn, val * cs) : fmaf(val, cs, -partner * sn);
        r *= 0.125f;                            // fold D^-0.5 into q
        qh[((size_t)(bb * H_ + h) * S_ + s) * D_ + d] = (__bf16)r;
      }
    }
}

// ---------------- 3) kv = lat[:, 64:] @ Wkvb, split K(+RoPE)/V, [b][kvh][s][d] -------------
__global__ __launch_bounds__(128) void k_kvproj(const __bf16* __restrict__ lat,
                                                const __bf16* __restrict__ wkvbT,
                                                __bf16* __restrict__ kh,
                                                __bf16* __restrict__ vh) {
  int w = threadIdx.x >> 5;
  int m0 = blockIdx.x * 128 + w * 32;
  int n0g = blockIdx.y * 64;
  v8f c[8] = {};
  v16bf a[2], b[4];
  a[0] = load_a_frag(lat + 64, 96, m0, 0);      // KVL = 32 -> single K step
  a[1] = load_a_frag(lat + 64, 96, m0 + 16, 0);
#pragma unroll
  for (int nt = 0; nt < 4; ++nt)
    b[nt] = load_bT_frag(wkvbT, KVL_, n0g + nt * 16, 0);
#pragma unroll
  for (int mi = 0; mi < 2; ++mi)
#pragma unroll
    for (int nt = 0; nt < 4; ++nt)
      c[mi * 4 + nt] = WMMA_BF16(a[mi], b[nt], c[mi * 4 + nt]);

  int lane = threadIdx.x & 31, half = lane >> 4, nm = lane & 15;
#pragma unroll
  for (int mi = 0; mi < 2; ++mi)
#pragma unroll
    for (int nt = 0; nt < 4; ++nt) {
      union { v8f v; float f[8]; } cc; cc.v = c[mi * 4 + nt];
#pragma unroll
      for (int j = 0; j < 8; ++j) {
        float val = cc.f[j];
        float partner = __shfl_xor(val, 2, 32); // K's RoPE pair is 2 columns away (k/v interleave)
        int colg = n0g + nt * 16 + nm;
        int kvh = colg >> 7;
        int d = (colg >> 1) & 63;
        int jf = d >> 1;
        int row = m0 + mi * 16 + j + half * 8;
        int s = row & (S_ - 1), bb = row >> 11;
        float theta = (float)s * __expf(-(float)jf * (LN10000 / 32.0f));
        float cs = __cosf(theta), sn = __sinf(theta);
        float rk = (d & 1) ? fmaf(partner, sn, val * cs) : fmaf(val, cs, -partner * sn);
        size_t o = ((size_t)(bb * KVH_ + kvh) * S_ + s) * D_ + d;
        if ((colg & 1) == 0) kh[o] = (__bf16)rk;  // t==0 -> key (rotated)
        else                 vh[o] = (__bf16)val; // t==1 -> value
      }
    }
}

// ---------------- 4) causal flash attention, 64 q-rows per block, 4 waves ------------------
__global__ __launch_bounds__(128) void k_attn(const __bf16* __restrict__ qh,
                                              const __bf16* __restrict__ kh,
                                              const __bf16* __restrict__ vh,
                                              __bf16* __restrict__ ao) {
  __shared__ __align__(16) __bf16 vT[64 * 72];        // V tile transposed [d][key], padded
  __shared__ __align__(16) __bf16 pbuf[4 * 16 * 64];  // per-wave P tile [row][key]

  int w = threadIdx.x >> 5;
  int lane = threadIdx.x & 31, half = lane >> 4, nm = lane & 15;
  int qt = blockIdx.x;
  int b = blockIdx.y >> 4, h = blockIdx.y & 15;
  int kvh = h >> 2;                                   // GQA: 4 query heads per kv head

  const __bf16* qbase = qh + ((size_t)(b * H_ + h) * S_ + qt * 64) * D_;
  v16bf aq0 = load_a_frag(qbase, D_, w * 16, 0);
  v16bf aq1 = load_a_frag(qbase, D_, w * 16, 32);

  float m_r[8], l_r[8];
  v8f acc[4] = {};
#pragma unroll
  for (int j = 0; j < 8; ++j) { m_r[j] = -1e30f; l_r[j] = 0.0f; }

  __bf16* pw = pbuf + w * 16 * 64;

  for (int kt = 0; kt <= qt; ++kt) {
    // stage V tile transposed into LDS
    __syncthreads();
    const __bf16* vsrc = vh + ((size_t)(b * KVH_ + kvh) * S_ + kt * 64) * D_;
    for (int idx = threadIdx.x; idx < 64 * 64; idx += 128)
      vT[(idx & 63) * 72 + (idx >> 6)] = vsrc[idx];
    __syncthreads();

    // scores: load all 8 K-fragments, then 8 WMMAs (waits amortized over the batch)
    const __bf16* kbase = kh + ((size_t)(b * KVH_ + kvh) * S_ + kt * 64) * D_;
    v16bf kb[8];
#pragma unroll
    for (int nt = 0; nt < 4; ++nt) {
      kb[nt]     = load_bT_frag(kbase, D_, nt * 16, 0);
      kb[4 + nt] = load_bT_frag(kbase, D_, nt * 16, 32);
    }
    v8f sf[4];
#pragma unroll
    for (int nt = 0; nt < 4; ++nt) {
      v8f cz = {};
      cz = WMMA_BF16(aq0, kb[nt], cz);
      cz = WMMA_BF16(aq1, kb[4 + nt], cz);
      sf[nt] = cz;
    }

    // causal mask (scale already folded into q), row max
    float tmax[8];
#pragma unroll
    for (int j = 0; j < 8; ++j) tmax[j] = -1e30f;
#pragma unroll
    for (int nt = 0; nt < 4; ++nt) {
      union { v8f v; float f[8]; } cc; cc.v = sf[nt];
#pragma unroll
      for (int j = 0; j < 8; ++j) {
        float vle = cc.f[j];
        if (kt == qt) {
          int key = kt * 64 + nt * 16 + nm;
          int qr  = qt * 64 + w * 16 + j + half * 8;
          if (key > qr) vle = -1e30f;
        }
        cc.f[j] = vle;
        tmax[j] = fmaxf(tmax[j], vle);
      }
      sf[nt] = cc.v;
    }
#pragma unroll
    for (int j = 0; j < 8; ++j) {
      tmax[j] = fmaxf(tmax[j], __shfl_xor(tmax[j], 1, 32));
      tmax[j] = fmaxf(tmax[j], __shfl_xor(tmax[j], 2, 32));
      tmax[j] = fmaxf(tmax[j], __shfl_xor(tmax[j], 4, 32));
      tmax[j] = fmaxf(tmax[j], __shfl_xor(tmax[j], 8, 32));
    }

    // online softmax rescale
    float corr[8];
#pragma unroll
    for (int j = 0; j < 8; ++j) {
      float mn = fmaxf(m_r[j], tmax[j]);
      corr[j] = __expf(m_r[j] - mn);
      m_r[j] = mn;
      l_r[j] *= corr[j];
    }
#pragma unroll
    for (int dt = 0; dt < 4; ++dt) {
      union { v8f v; float f[8]; } cc; cc.v = acc[dt];
#pragma unroll
      for (int j = 0; j < 8; ++j) cc.f[j] *= corr[j];
      acc[dt] = cc.v;
    }

    // P = exp(S - m), accumulate l, round-trip through LDS to A-fragment layout
#pragma unroll
    for (int nt = 0; nt < 4; ++nt) {
      union { v8f v; float f[8]; } cc; cc.v = sf[nt];
#pragma unroll
      for (int j = 0; j < 8; ++j) {
        float pe = __expf(cc.f[j] - m_r[j]);
        l_r[j] += pe;
        pw[(j + half * 8) * 64 + nt * 16 + nm] = (__bf16)pe;
      }
    }
    __syncthreads();

    // O += P(16x64) @ V(64x64); load all 8 V^T fragments + 2 P fragments, then 8 WMMAs
    v16bf vb[8];
#pragma unroll
    for (int dt = 0; dt < 4; ++dt) {
      vb[dt]     = load_bT_frag(vT, 72, dt * 16, 0);
      vb[4 + dt] = load_bT_frag(vT, 72, dt * 16, 32);
    }
    v16bf pa0 = load_a_frag(pw, 64, 0, 0);
    v16bf pa1 = load_a_frag(pw, 64, 0, 32);
#pragma unroll
    for (int dt = 0; dt < 4; ++dt) {
      acc[dt] = WMMA_BF16(pa0, vb[dt], acc[dt]);
      acc[dt] = WMMA_BF16(pa1, vb[4 + dt], acc[dt]);
    }
  }

  // finalize: divide by l, write [b][s][h*64+d]
#pragma unroll
  for (int dt = 0; dt < 4; ++dt) {
    union { v8f v; float f[8]; } cc; cc.v = acc[dt];
#pragma unroll
    for (int j = 0; j < 8; ++j) {
      int s = qt * 64 + w * 16 + j + half * 8;
      int d = dt * 16 + nm;
      ao[(size_t)(b * S_ + s) * E_ + h * 64 + d] = (__bf16)(cc.f[j] / l_r[j]);
    }
  }
}

// ---------------- 5) out = ao @ Wout + bout, fully software-pipelined ----------------------
__global__ __launch_bounds__(128) void k_outproj(const __bf16* __restrict__ ao,
                                                 const __bf16* __restrict__ woutT,
                                                 const float* __restrict__ bout,
                                                 float* __restrict__ out) {
  int w = threadIdx.x >> 5;
  int m0 = blockIdx.x * 128 + w * 32;
  int n0g = blockIdx.y * 64;
  v8f c[8] = {};
  v16bf a[2], b[4], an[2], bn[4];
  a[0] = load_a_frag(ao, E_, m0, 0);
  a[1] = load_a_frag(ao, E_, m0 + 16, 0);
#pragma unroll
  for (int nt = 0; nt < 4; ++nt) b[nt] = load_bT_frag(woutT, E_, n0g + nt * 16, 0);

  for (int k = 32; k < E_; k += 32) {
    // prefetch next K-step while current WMMAs execute
    an[0] = load_a_frag(ao, E_, m0, k);
    an[1] = load_a_frag(ao, E_, m0 + 16, k);
#pragma unroll
    for (int nt = 0; nt < 4; ++nt) bn[nt] = load_bT_frag(woutT, E_, n0g + nt * 16, k);
#pragma unroll
    for (int mi = 0; mi < 2; ++mi)
#pragma unroll
      for (int nt = 0; nt < 4; ++nt)
        c[mi * 4 + nt] = WMMA_BF16(a[mi], b[nt], c[mi * 4 + nt]);
    a[0] = an[0]; a[1] = an[1];
#pragma unroll
    for (int nt = 0; nt < 4; ++nt) b[nt] = bn[nt];
  }
#pragma unroll
  for (int mi = 0; mi < 2; ++mi)
#pragma unroll
    for (int nt = 0; nt < 4; ++nt)
      c[mi * 4 + nt] = WMMA_BF16(a[mi], b[nt], c[mi * 4 + nt]);

  int lane = threadIdx.x & 31, half = lane >> 4, nm = lane & 15;
#pragma unroll
  for (int mi = 0; mi < 2; ++mi)
#pragma unroll
    for (int nt = 0; nt < 4; ++nt) {
      union { v8f v; float f[8]; } cc; cc.v = c[mi * 4 + nt];
#pragma unroll
      for (int j = 0; j < 8; ++j) {
        int row = m0 + mi * 16 + j + half * 8;
        int col = n0g + nt * 16 + nm;
        out[(size_t)row * E_ + col] = cc.f[j] + bout[col];
      }
    }
}

// ---------------- launcher -----------------------------------------------------------------
extern "C" void kernel_launch(void* const* d_in, const int* in_sizes, int n_in,
                              void* d_out, int out_size, void* d_ws, size_t ws_size,
                              hipStream_t stream) {
  (void)in_sizes; (void)n_in; (void)out_size; (void)ws_size;
  const float* x    = (const float*)d_in[0];
  const float* Wqa  = (const float*)d_in[1];
  const float* Wqb  = (const float*)d_in[2];
  const float* Wkva = (const float*)d_in[3];
  const float* Wkvb = (const float*)d_in[4];
  const float* Wout = (const float*)d_in[5];
  const float* bout = (const float*)d_in[6];
  float* out = (float*)d_out;

  char* p = (char*)d_ws;
  auto alloc = [&](size_t bytes) -> char* {
    char* r = p; p += (bytes + 255) & ~(size_t)255; return r;
  };
  __bf16* xb    = (__bf16*)alloc((size_t)BS_ * E_ * 2);
  __bf16* wqaT  = (__bf16*)alloc((size_t)E_ * QL_ * 2);
  __bf16* wqbT  = (__bf16*)alloc((size_t)QL_ * E_ * 2);
  __bf16* wkvaT = (__bf16*)alloc((size_t)E_ * KVL_ * 2);
  __bf16* wkvbT = (__bf16*)alloc((size_t)KVL_ * 512 * 2);
  __bf16* woutT = (__bf16*)alloc((size_t)E_ * E_ * 2);
  __bf16* lat   = (__bf16*)alloc((size_t)BS_ * 96 * 2);
  __bf16* qh    = (__bf16*)alloc((size_t)BS_ * E_ * 2);
  __bf16* kh    = (__bf16*)alloc((size_t)BS_ * KVH_ * D_ * 2);
  __bf16* vh    = (__bf16*)alloc((size_t)BS_ * KVH_ * D_ * 2);
  __bf16* ao    = (__bf16*)alloc((size_t)BS_ * E_ * 2);

  int nx8 = (BS_ * E_) / 8;
  k_cast8<<<(nx8 + 255) / 256, 256, 0, stream>>>(x, xb, nx8);
  k_cast_transpose<<<(E_ * QL_ + 255) / 256, 256, 0, stream>>>(Wqa, wqaT, E_, QL_);
  k_cast_transpose<<<(QL_ * E_ + 255) / 256, 256, 0, stream>>>(Wqb, wqbT, QL_, E_);
  k_cast_transpose<<<(E_ * KVL_ + 255) / 256, 256, 0, stream>>>(Wkva, wkvaT, E_, KVL_);
  k_cast_transpose<<<(KVL_ * 512 + 255) / 256, 256, 0, stream>>>(Wkvb, wkvbT, KVL_, 512);
  k_cast_transpose<<<(E_ * E_ + 255) / 256, 256, 0, stream>>>(Wout, woutT, E_, E_);

  k_lat<<<BS_ / 128, 128, 0, stream>>>(xb, wqaT, wkvaT, lat);
  k_qproj<<<dim3(BS_ / 128, H_), 128, 0, stream>>>(lat, wqbT, qh);
  k_kvproj<<<dim3(BS_ / 128, 8), 128, 0, stream>>>(lat, wkvbT, kh, vh);
  k_attn<<<dim3(S_ / 64, B_ * H_), 128, 0, stream>>>(qh, kh, vh, ao);
  k_outproj<<<dim3(BS_ / 128, E_ / 64), 128, 0, stream>>>(ao, woutT, bout, out);
}
